// EncoderLayer_42597485642367
// MI455X (gfx1250) — matrix operand order repaired and checked by reading, not compiled, not run on hardware
//
#include <hip/hip_runtime.h>
#include <hip/hip_bf16.h>

typedef __bf16 bf16;
typedef __attribute__((ext_vector_type(16))) __bf16 bf16x16;
typedef __attribute__((ext_vector_type(8)))  __bf16 bf16x8;
typedef __attribute__((ext_vector_type(8)))  float  f32x8;

// ---------------------------------------------------------------------------
// Batched bf16 WMMA GEMM, B always stored transposed (N x K row-major):
//   C[m,n] = alpha * sum_k A[m,k] * Bt[n,k]  (+bias[n]) (+ReLU)
// One wave computes a 64(M) x 64(N) macro-tile = 4x4 grid of 16x16 WMMA tiles:
// 16 v_wmma per k-step against 16 b128 fragment loads (1 load : 1 WMMA).
// Fragment layouts follow CDNA5 ISA 7.12.2 (wave32):
//   A 16x32 bf16: lanes 0-15 hold row M=lane, K={0..7,16..23}; lanes 16-31 K={8..15,24..31}
//   B 32x16 bf16: lanes 0-15 hold col N=lane, K={0..7,16..23}; lanes 16-31 K={8..15,24..31}
//   C 16x16 f32 : VGPR i -> M=(lane>=16?8:0)+i, N=lane%16
// ---------------------------------------------------------------------------
template <bool BIAS, bool RELU, bool OBF>
__global__ __launch_bounds__(32) void gemm_bt_wmma(
    const bf16* __restrict__ A, long long lda,
    const bf16* __restrict__ Bt, long long ldb,
    const float* __restrict__ bias,
    void* __restrict__ Cv, long long ldc,
    int Kk, int Hdim,
    long long aOffB, long long aOffH,
    long long bOffB, long long bOffH,
    long long cOffB, long long cOffH,
    float alpha)
{
  const int z  = blockIdx.z;
  const int bb = z / Hdim;
  const int hh = z % Hdim;

  const bf16* Ab = A  + (long long)bb * aOffB + (long long)hh * aOffH;
  const bf16* Bb = Bt + (long long)bb * bOffB + (long long)hh * bOffH;

  const int lane = threadIdx.x & 31;
  const int kh   = lane >> 4;       // K-half selector for this lane
  const int l16  = lane & 15;

  const int m0 = blockIdx.x * 64 + l16;   // A row of first M-tile for this lane
  const int n0 = blockIdx.y * 64 + l16;   // B col of first N-tile for this lane

  const bf16* ap[4];
  const bf16* bp[4];
#pragma unroll
  for (int mt = 0; mt < 4; ++mt) ap[mt] = Ab + (long long)(m0 + mt * 16) * lda + kh * 8;
#pragma unroll
  for (int nt = 0; nt < 4; ++nt) bp[nt] = Bb + (long long)(n0 + nt * 16) * ldb + kh * 8;

  f32x8 acc[4][4] = {};

  union Frag { bf16x16 v; bf16x8 h[2]; };

  for (int k0 = 0; k0 < Kk; k0 += 32) {
    Frag a[4], b[4];
    // A fragment: elements 0..7 = K {k0+kh*8 .. +7}, elements 8..15 = +16
#pragma unroll
    for (int mt = 0; mt < 4; ++mt) {
      a[mt].h[0] = *(const bf16x8*)(ap[mt] + k0);
      a[mt].h[1] = *(const bf16x8*)(ap[mt] + k0 + 16);
    }
    // B fragment: elements 0..7 = K {k0+kh*8 .. +7}, elements 8..15 = +16
#pragma unroll
    for (int nt = 0; nt < 4; ++nt) {
      b[nt].h[0] = *(const bf16x8*)(bp[nt] + k0);
      b[nt].h[1] = *(const bf16x8*)(bp[nt] + k0 + 16);
    }

    if (k0 + 32 < Kk) {
      __builtin_prefetch((const void*)(bp[0] + k0 + 32), 0, 1);
      __builtin_prefetch((const void*)(bp[2] + k0 + 32), 0, 1);
    }

#pragma unroll
    for (int mt = 0; mt < 4; ++mt)
#pragma unroll
      for (int nt = 0; nt < 4; ++nt)
        acc[mt][nt] = __builtin_amdgcn_wmma_f32_16x16x32_bf16(
            false, a[mt].v, false, b[nt].v, (short)0, acc[mt][nt], false, false);
  }

  const long long cbase = (long long)bb * cOffB + (long long)hh * cOffH;
#pragma unroll
  for (int mt = 0; mt < 4; ++mt) {
    const int mb = blockIdx.x * 64 + mt * 16 + kh * 8;
#pragma unroll
    for (int nt = 0; nt < 4; ++nt) {
      const int cn = blockIdx.y * 64 + nt * 16 + l16;
      const float badd = BIAS ? bias[cn] : 0.f;
#pragma unroll
      for (int i = 0; i < 8; ++i) {
        float v = acc[mt][nt][i] * alpha + badd;
        if (RELU) v = v > 0.f ? v : 0.f;
        const long long cidx = cbase + (long long)(mb + i) * ldc + cn;
        if (OBF) ((bf16*)Cv)[cidx] = (bf16)v;
        else     ((float*)Cv)[cidx] = v;
      }
    }
  }
}

// ---------------------------------------------------------------------------
// Elementwise / reduction helper kernels
// ---------------------------------------------------------------------------
__global__ __launch_bounds__(256) void f32_to_bf16_k(const float* __restrict__ in,
                                                     bf16* __restrict__ out, long long n) {
  for (long long i = blockIdx.x * 256LL + threadIdx.x; i < n; i += (long long)gridDim.x * 256LL)
    out[i] = (bf16)in[i];
}

// Wt[n*K + k] = (bf16) W[k*N + n]   (convert + transpose, coalesced on the read)
__global__ __launch_bounds__(256) void cvt_tr_k(const float* __restrict__ W,
                                                bf16* __restrict__ Wt, int K, int N) {
  const long long n = (long long)K * N;
  for (long long i = blockIdx.x * 256LL + threadIdx.x; i < n; i += (long long)gridDim.x * 256LL) {
    const int kk = (int)(i / N);
    const int nn = (int)(i % N);
    Wt[(long long)nn * K + kk] = (bf16)W[i];
  }
}

// Vt[((b*H+h)*dh + d)*L + l] = V[(b*L+l)*D + h*dh + d]
__global__ __launch_bounds__(256) void transpose_v_k(const bf16* __restrict__ Vb,
                                                     bf16* __restrict__ Vt,
                                                     int B_, int H_, int dh, int L) {
  const int D = H_ * dh;
  const long long n = (long long)B_ * L * D;
  for (long long i = blockIdx.x * 256LL + threadIdx.x; i < n; i += (long long)gridDim.x * 256LL) {
    const int d2 = (int)(i % D);
    const long long bl = i / D;
    const int l = (int)(bl % L);
    const int b = (int)(bl / L);
    const int h = d2 / dh;
    const int d = d2 % dh;
    Vt[(((long long)b * H_ + h) * dh + d) * L + l] = Vb[i];
  }
}

// M = max_s(ss) - mean_s(ss) over SK=128 sampled scores; one block (128 thr) per row
__global__ __launch_bounds__(128) void mmetric_k(const float* __restrict__ ss,
                                                 float* __restrict__ Mout, int SK) {
  __shared__ float smax[128];
  __shared__ float ssum[128];
  const int t = threadIdx.x;
  const float v = ss[(long long)blockIdx.x * SK + t];
  smax[t] = v; ssum[t] = v; __syncthreads();
  for (int s = 64; s > 0; s >>= 1) {
    if (t < s) { smax[t] = fmaxf(smax[t], smax[t + s]); ssum[t] += ssum[t + s]; }
    __syncthreads();
  }
  if (t == 0) Mout[blockIdx.x] = smax[0] - ssum[0] / (float)SK;
}

// Top-U selection per (b,h): iterative argmax over an LDS copy of the L scores
__global__ __launch_bounds__(256) void topk_k(const float* __restrict__ Mm,
                                              int* __restrict__ idxOut, int L, int U) {
  __shared__ float cache[4096];
  __shared__ float rv[256];
  __shared__ int   ri[256];
  const int t = threadIdx.x;
  const long long base = (long long)blockIdx.x * L;
  for (int i = t; i < L; i += 256) cache[i] = Mm[base + i];
  __syncthreads();
  for (int u = 0; u < U; ++u) {
    float bv = -__builtin_inff(); int bi = 0x7fffffff;
    for (int i = t; i < L; i += 256) {
      const float c = cache[i];
      if (c > bv) { bv = c; bi = i; }
    }
    rv[t] = bv; ri[t] = bi; __syncthreads();
    for (int s = 128; s > 0; s >>= 1) {
      if (t < s) {
        if (rv[t + s] > rv[t] || (rv[t + s] == rv[t] && ri[t + s] < ri[t])) {
          rv[t] = rv[t + s]; ri[t] = ri[t + s];
        }
      }
      __syncthreads();
    }
    if (t == 0) { idxOut[(long long)blockIdx.x * U + u] = ri[0]; cache[ri[0]] = -__builtin_inff(); }
    __syncthreads();
  }
}

// K_samp[b,h,s,d] = K[b, samp[s], h, d]
__global__ __launch_bounds__(256) void gather_ksamp_k(const bf16* __restrict__ Kb,
                                                      const long long* __restrict__ samp,
                                                      bf16* __restrict__ Ks,
                                                      int B_, int H_, int SK, int dh, int L, int D) {
  const long long n = (long long)B_ * H_ * SK * dh;
  for (long long i = blockIdx.x * 256LL + threadIdx.x; i < n; i += (long long)gridDim.x * 256LL) {
    const int d = (int)(i % dh);
    const int s = (int)((i / dh) % SK);
    const int h = (int)((i / ((long long)dh * SK)) % H_);
    const int b = (int)(i / ((long long)dh * SK * H_));
    const long long l = samp[s];
    Ks[i] = Kb[((long long)b * L + l) * D + (long long)h * dh + d];
  }
}

// Q_top[b,h,u,d] = Q[b, top[b,h,u], h, d]
__global__ __launch_bounds__(256) void gather_qtop_k(const bf16* __restrict__ Qb,
                                                     const int* __restrict__ topidx,
                                                     bf16* __restrict__ Qt,
                                                     int B_, int H_, int U, int dh, int L, int D) {
  const long long n = (long long)B_ * H_ * U * dh;
  for (long long i = blockIdx.x * 256LL + threadIdx.x; i < n; i += (long long)gridDim.x * 256LL) {
    const int d = (int)(i % dh);
    const int u = (int)((i / dh) % U);
    const int h = (int)((i / ((long long)dh * U)) % H_);
    const int b = (int)(i / ((long long)dh * U * H_));
    const int l = topidx[((long long)b * H_ + h) * U + u];
    Qt[i] = Qb[((long long)b * L + l) * D + (long long)h * dh + d];
  }
}

// Row softmax over L columns (scale applied pre-exp), output bf16
__global__ __launch_bounds__(256) void softmax_k(const float* __restrict__ S,
                                                 bf16* __restrict__ P, int L, float scale) {
  __shared__ float red[256];
  const long long base = (long long)blockIdx.x * L;
  const int t = threadIdx.x;
  float mx = -__builtin_inff();
  for (int i = t; i < L; i += 256) mx = fmaxf(mx, S[base + i] * scale);
  red[t] = mx; __syncthreads();
  for (int s = 128; s > 0; s >>= 1) { if (t < s) red[t] = fmaxf(red[t], red[t + s]); __syncthreads(); }
  mx = red[0]; __syncthreads();
  float sum = 0.f;
  for (int i = t; i < L; i += 256) sum += __expf(S[base + i] * scale - mx);
  red[t] = sum; __syncthreads();
  for (int s = 128; s > 0; s >>= 1) { if (t < s) red[t] += red[t + s]; __syncthreads(); }
  const float inv = 1.f / red[0];
  for (int i = t; i < L; i += 256) P[base + i] = (bf16)(__expf(S[base + i] * scale - mx) * inv);
}

// meanV[b,h,d] = mean over L of V[b,l,h,d]; one block per (b,h,d)
__global__ __launch_bounds__(256) void meanv_k(const bf16* __restrict__ Vb,
                                               bf16* __restrict__ mv,
                                               int H_, int dh, int L, int D) {
  __shared__ float red[256];
  const int id = blockIdx.x;
  const int d = id % dh;
  const int h = (id / dh) % H_;
  const int b = id / (dh * H_);
  const int t = threadIdx.x;
  float s = 0.f;
  const bf16* p = Vb + (long long)b * L * D + (long long)h * dh + d;
  for (int l = t; l < L; l += 256) s += (float)p[(long long)l * D];
  red[t] = s; __syncthreads();
  for (int k = 128; k > 0; k >>= 1) { if (t < k) red[t] += red[t + k]; __syncthreads(); }
  if (t == 0) mv[id] = (bf16)(red[0] / (float)L);
}

// ctx[b,l,h,d] = meanV[b,h,d]  (broadcast)
__global__ __launch_bounds__(256) void fill_ctx_k(const bf16* __restrict__ mv,
                                                  bf16* __restrict__ ctx,
                                                  long long n, int L, int H_, int dh) {
  for (long long i = blockIdx.x * 256LL + threadIdx.x; i < n; i += (long long)gridDim.x * 256LL) {
    const int d = (int)(i % dh);
    const int h = (int)((i / dh) % H_);
    const long long bl = i / ((long long)dh * H_);
    const int b = (int)(bl / L);
    ctx[i] = mv[((long long)b * H_ + h) * dh + d];
  }
}

// ctx[b, top[b,h,u], h, d] = ctx_top[b,h,u,d]
__global__ __launch_bounds__(256) void scatter_ctx_k(const bf16* __restrict__ ctxtop,
                                                     const int* __restrict__ topidx,
                                                     bf16* __restrict__ ctx,
                                                     int B_, int H_, int U, int dh, int L) {
  const long long n = (long long)B_ * H_ * U * dh;
  const int D = H_ * dh;
  for (long long i = blockIdx.x * 256LL + threadIdx.x; i < n; i += (long long)gridDim.x * 256LL) {
    const int d = (int)(i % dh);
    const int u = (int)((i / dh) % U);
    const int h = (int)((i / ((long long)dh * U)) % H_);
    const int b = (int)(i / ((long long)dh * U * H_));
    const int l = topidx[((long long)b * H_ + h) * U + u];
    ctx[((long long)b * L + l) * D + (long long)h * dh + d] = ctxtop[i];
  }
}

// out = LayerNorm(X + Y) * g + beta ; optional f32 and bf16 outputs. One block per row.
__global__ __launch_bounds__(256) void add_ln_k(const float* __restrict__ X,
                                                const float* __restrict__ Y,
                                                const float* __restrict__ g,
                                                const float* __restrict__ beta,
                                                float* __restrict__ outF,
                                                bf16* __restrict__ outB, int D) {
  __shared__ float red[256];
  const long long base = (long long)blockIdx.x * D;
  const int t = threadIdx.x;
  float s = 0.f;
  for (int i = t; i < D; i += 256) s += X[base + i] + Y[base + i];
  red[t] = s; __syncthreads();
  for (int k = 128; k > 0; k >>= 1) { if (t < k) red[t] += red[t + k]; __syncthreads(); }
  const float mean = red[0] / (float)D; __syncthreads();
  float vs = 0.f;
  for (int i = t; i < D; i += 256) { const float v = X[base + i] + Y[base + i] - mean; vs += v * v; }
  red[t] = vs; __syncthreads();
  for (int k = 128; k > 0; k >>= 1) { if (t < k) red[t] += red[t + k]; __syncthreads(); }
  const float rstd = rsqrtf(red[0] / (float)D + 1e-6f);
  for (int i = t; i < D; i += 256) {
    const float v = (X[base + i] + Y[base + i] - mean) * rstd * g[i] + beta[i];
    if (outF) outF[base + i] = v;
    if (outB) outB[base + i] = (bf16)v;
  }
}

// ---------------------------------------------------------------------------
// Host-side orchestration
// ---------------------------------------------------------------------------
extern "C" void kernel_launch(void* const* d_in, const int* in_sizes, int n_in,
                              void* d_out, int out_size, void* d_ws, size_t ws_size,
                              hipStream_t stream) {
  (void)in_sizes; (void)n_in; (void)out_size; (void)ws_size;

  const int Bn = 8, L = 4096, D = 512, Dff = 2048, SK = 128, H = 8, dh = 64, U = 128;
  const int N = Bn * L;                 // 32768 rows
  const float scale = 0.125f;           // 1/sqrt(dh)

  const float*      x    = (const float*)d_in[0];
  const float*      Wq   = (const float*)d_in[1];
  const float*      bq   = (const float*)d_in[2];
  const float*      Wk   = (const float*)d_in[3];
  const float*      bk   = (const float*)d_in[4];
  const float*      Wv   = (const float*)d_in[5];
  const float*      bv   = (const float*)d_in[6];
  const float*      Wo   = (const float*)d_in[7];
  const float*      bo   = (const float*)d_in[8];
  const float*      g1   = (const float*)d_in[9];
  const float*      b1   = (const float*)d_in[10];
  const float*      W1f  = (const float*)d_in[11];
  const float*      b1f  = (const float*)d_in[12];
  const float*      W2f  = (const float*)d_in[13];
  const float*      b2f  = (const float*)d_in[14];
  const float*      g2   = (const float*)d_in[15];
  const float*      b2v  = (const float*)d_in[16];
  const long long*  samp = (const long long*)d_in[17];
  float* out = (float*)d_out;

  // Workspace carve-out (256B aligned)
  char* w = (char*)d_ws;
  auto alloc = [&](size_t bytes) -> void* {
    void* p = (void*)w;
    w += (bytes + 255) & ~(size_t)255;
    return p;
  };
  bf16* xb    = (bf16*)alloc((size_t)N * D * 2);
  bf16* Wqt   = (bf16*)alloc((size_t)D * D * 2);     // transposed (N x K)
  bf16* Wkt   = (bf16*)alloc((size_t)D * D * 2);
  bf16* Wvt   = (bf16*)alloc((size_t)D * D * 2);
  bf16* Wot   = (bf16*)alloc((size_t)D * D * 2);
  bf16* W1t   = (bf16*)alloc((size_t)D * Dff * 2);   // (Dff x D)
  bf16* W2t   = (bf16*)alloc((size_t)Dff * D * 2);   // (D x Dff)
  bf16* Qb    = (bf16*)alloc((size_t)N * D * 2);
  bf16* Kb    = (bf16*)alloc((size_t)N * D * 2);
  bf16* Vb    = (bf16*)alloc((size_t)N * D * 2);
  bf16* Vt    = (bf16*)alloc((size_t)N * D * 2);     // (B,H,dh,L)
  bf16* Ks    = (bf16*)alloc((size_t)Bn * H * SK * dh * 2);
  float* ssamp = (float*)alloc((size_t)Bn * H * L * SK * 4);
  float* Mmet  = (float*)alloc((size_t)Bn * H * L * 4);
  int*   topi  = (int*)alloc((size_t)Bn * H * U * 4);
  bf16* Qt    = (bf16*)alloc((size_t)Bn * H * U * dh * 2);
  float* scor  = (float*)alloc((size_t)Bn * H * U * L * 4);
  bf16* attnP = (bf16*)alloc((size_t)Bn * H * U * L * 2);
  bf16* ctxT  = (bf16*)alloc((size_t)Bn * H * U * dh * 2);
  bf16* mv    = (bf16*)alloc((size_t)Bn * H * dh * 2);
  bf16* ctx   = (bf16*)alloc((size_t)N * D * 2);
  float* aout  = (float*)alloc((size_t)N * D * 4);
  float* x1f   = (float*)alloc((size_t)N * D * 4);
  bf16* x1b   = (bf16*)alloc((size_t)N * D * 2);
  bf16* hbuf  = (bf16*)alloc((size_t)N * Dff * 2);
  float* ffn   = (float*)alloc((size_t)N * D * 4);

  // 1) bf16 conversions (+ weight transposes so every GEMM uses contiguous B^T loads)
  f32_to_bf16_k<<<dim3(4096), 256, 0, stream>>>(x, xb, (long long)N * D);
  cvt_tr_k<<<dim3(512),  256, 0, stream>>>(Wq,  Wqt, D, D);
  cvt_tr_k<<<dim3(512),  256, 0, stream>>>(Wk,  Wkt, D, D);
  cvt_tr_k<<<dim3(512),  256, 0, stream>>>(Wv,  Wvt, D, D);
  cvt_tr_k<<<dim3(512),  256, 0, stream>>>(Wo,  Wot, D, D);
  cvt_tr_k<<<dim3(1024), 256, 0, stream>>>(W1f, W1t, D, Dff);
  cvt_tr_k<<<dim3(1024), 256, 0, stream>>>(W2f, W2t, Dff, D);

  // 2) Q/K/V projections: (N x D) = (N x D)(D x D) + bias  -> bf16
  {
    dim3 g(N / 64, D / 64, 1);
    gemm_bt_wmma<true, false, true><<<g, 32, 0, stream>>>(
        xb, D, Wqt, D, bq, Qb, D, D, 1, 0, 0, 0, 0, 0, 0, 1.f);
    gemm_bt_wmma<true, false, true><<<g, 32, 0, stream>>>(
        xb, D, Wkt, D, bk, Kb, D, D, 1, 0, 0, 0, 0, 0, 0, 1.f);
    gemm_bt_wmma<true, false, true><<<g, 32, 0, stream>>>(
        xb, D, Wvt, D, bv, Vb, D, D, 1, 0, 0, 0, 0, 0, 0, 1.f);
  }

  // 3) K_samp gather + V transpose
  gather_ksamp_k<<<dim3(512), 256, 0, stream>>>(Kb, samp, Ks, Bn, H, SK, dh, L, D);
  transpose_v_k<<<dim3(4096), 256, 0, stream>>>(Vb, Vt, Bn, H, dh, L);

  // 4) s_samp[b,h,l,s] = Q . K_samp^T  (batched over b,h; K = dh; Ks already N x K)
  {
    dim3 g(L / 64, SK / 64, Bn * H);
    gemm_bt_wmma<false, false, false><<<g, 32, 0, stream>>>(
        Qb, D, Ks, dh, nullptr, ssamp, SK, dh, H,
        (long long)L * D, (long long)dh,
        (long long)H * SK * dh, (long long)SK * dh,
        (long long)H * L * SK, (long long)L * SK,
        1.f);
  }

  // 5) M = max - mean over the SK sampled scores
  mmetric_k<<<dim3(Bn * H * L), 128, 0, stream>>>(ssamp, Mmet, SK);

  // 6) top-U per (b,h)
  topk_k<<<dim3(Bn * H), 256, 0, stream>>>(Mmet, topi, L, U);

  // 7) gather Q_top
  gather_qtop_k<<<dim3(512), 256, 0, stream>>>(Qb, topi, Qt, Bn, H, U, dh, L, D);

  // 8) scores[b,h,u,l] = scale * Q_top . K^T  (K rows are N x K for B^T)
  {
    dim3 g(U / 64, L / 64, Bn * H);
    gemm_bt_wmma<false, false, false><<<g, 32, 0, stream>>>(
        Qt, dh, Kb, D, nullptr, scor, L, dh, H,
        (long long)H * U * dh, (long long)U * dh,
        (long long)L * D, (long long)dh,
        (long long)H * U * L, (long long)U * L,
        scale);
  }

  // 9) softmax over L -> bf16 attn probs
  softmax_k<<<dim3(Bn * H * U), 256, 0, stream>>>(scor, attnP, L, 1.f);

  // 10) ctx_top[b,h,u,d] = attn . V   (K = L; B^T = Vt (dh x L))
  {
    dim3 g(U / 64, dh / 64, Bn * H);
    gemm_bt_wmma<false, false, true><<<g, 32, 0, stream>>>(
        attnP, L, Vt, L, nullptr, ctxT, dh, L, H,
        (long long)H * U * L, (long long)U * L,
        (long long)H * dh * L, (long long)dh * L,
        (long long)H * U * dh, (long long)U * dh,
        1.f);
  }

  // 11) meanV, 12) broadcast into ctx, 13) scatter top contexts
  meanv_k<<<dim3(Bn * H * dh), 256, 0, stream>>>(Vb, mv, H, dh, L, D);
  fill_ctx_k<<<dim3(4096), 256, 0, stream>>>(mv, ctx, (long long)N * D, L, H, dh);
  scatter_ctx_k<<<dim3(512), 256, 0, stream>>>(ctxT, topi, ctx, Bn, H, U, dh, L);

  // 14) attn_out = ctx @ Wo + bo  -> f32
  {
    dim3 g(N / 64, D / 64, 1);
    gemm_bt_wmma<true, false, false><<<g, 32, 0, stream>>>(
        ctx, D, Wot, D, bo, aout, D, D, 1, 0, 0, 0, 0, 0, 0, 1.f);
  }

  // 15) x1 = LN(x + attn_out) -> f32 + bf16
  add_ln_k<<<dim3(N), 256, 0, stream>>>(x, aout, g1, b1, x1f, x1b, D);

  // 16) h = ReLU(x1 @ W1f + b1f) -> bf16
  {
    dim3 g(N / 64, Dff / 64, 1);
    gemm_bt_wmma<true, true, true><<<g, 32, 0, stream>>>(
        x1b, D, W1t, D, b1f, hbuf, Dff, D, 1, 0, 0, 0, 0, 0, 0, 1.f);
  }

  // 17) ffn = h @ W2f + b2f -> f32
  {
    dim3 g(N / 64, D / 64, 1);
    gemm_bt_wmma<true, false, false><<<g, 32, 0, stream>>>(
        hbuf, Dff, W2t, Dff, b2f, ffn, D, Dff, 1, 0, 0, 0, 0, 0, 0, 1.f);
  }

  // 18) out = LN(x1 + ffn) -> f32
  add_ln_k<<<dim3(N), 256, 0, stream>>>(x1f, ffn, g2, b2v, out, nullptr, D);
}
